// PatchedVisionExpertAttention_65317862638205
// MI455X (gfx1250) — matrix operand order repaired and checked by reading, not compiled, not run on hardware
//
#include <hip/hip_runtime.h>
#include <hip/hip_bf16.h>
#include <stdint.h>

// Problem constants (from reference)
#define SEQ   2048
#define DM    4096
#define NH    32
#define HDIM  128
#define QKVN  (3 * DM)

typedef __bf16 bf16t;
typedef bf16t v16bf __attribute__((ext_vector_type(16)));
typedef float v8f  __attribute__((ext_vector_type(8)));

union FragU { uint4 u[2]; v16bf v; };

__device__ __forceinline__ unsigned short f2bf(float f) {
  union { float f; unsigned u; } x; x.f = f;
  return (unsigned short)((x.u + 0x7FFFu + ((x.u >> 16) & 1u)) >> 16);
}
__device__ __forceinline__ float bf2f(unsigned short h) {
  union { unsigned u; float f; } x; x.u = ((unsigned)h) << 16; return x.f;
}

// ---- WMMA fragment loaders (wave32, 16x16x32 bf16) -------------------------
// A fragment (16 rows x 32 K), ISA-documented swizzle:
//   lane&15 = row, half-wave selects K offset {0,8}; VGPR0..3: K=half*8+0..7,
//   VGPR4..7: K=16+half*8+0..7  -> two 16B loads per lane.
template <typename T>
__device__ __forceinline__ v16bf load_frag_a(const T* base, int row0, int ld,
                                             int k0, int lane) {
  int r  = row0 + (lane & 15);
  int kh = (lane >> 4) << 3;
  const T* p = base + (size_t)r * ld + k0 + kh;
  FragU f;
  f.u[0] = *(const uint4*)(p);
  f.u[1] = *(const uint4*)(p + 16);
  return f.v;
}
// B fragment (32 K x 16 cols): lane&15 = col, same K swizzle as A (half*8+0..7, 16+half*8+0..7).
template <typename T>
__device__ __forceinline__ v16bf load_frag_b(const T* base, int col0, int ld,
                                             int k0, int lane) {
  int c  = col0 + (lane & 15);
  int kh = (lane >> 4) << 3;
  const T* p = base + (size_t)c * ld + k0 + kh;
  FragU f;
  f.u[0] = *(const uint4*)(p);
  f.u[1] = *(const uint4*)(p + 16);
  return f.v;
}

__device__ __forceinline__ v8f wmma_bf16(v16bf a, v16bf b, v8f c) {
  return __builtin_amdgcn_wmma_f32_16x16x32_bf16(false, a, false, b, (short)0,
                                                 c, false, false);
}

// reductions within the 16-lane half-wave groups (wave32)
__device__ __forceinline__ float red_max16(float x) {
  x = fmaxf(x, __shfl_xor(x, 1));
  x = fmaxf(x, __shfl_xor(x, 2));
  x = fmaxf(x, __shfl_xor(x, 4));
  x = fmaxf(x, __shfl_xor(x, 8));
  return x;
}
__device__ __forceinline__ float red_sum16(float x) {
  x += __shfl_xor(x, 1);
  x += __shfl_xor(x, 2);
  x += __shfl_xor(x, 4);
  x += __shfl_xor(x, 8);
  return x;
}

// ---- helper kernels --------------------------------------------------------
__global__ void f32_to_bf16_kernel(const float* __restrict__ in,
                                   unsigned short* __restrict__ out,
                                   long long n) {
  long long i = (long long)blockIdx.x * blockDim.x + threadIdx.x;
  long long stride = (long long)gridDim.x * blockDim.x;
  for (; i < n; i += stride) out[i] = f2bf(in[i]);
}

__global__ void build_mask_kernel(const int* __restrict__ tt,
                                  int* __restrict__ mask) {
  int s = blockIdx.x * blockDim.x + threadIdx.x;
  if (s < SEQ)
    mask[s] = (s < SEQ - 1 && tt[s] == 1 && tt[s + 1] == 1) ? 1 : 0;
}

// RoPE + split into head-major q/k/v (bf16). Softmax scale folded into q.
__global__ void rope_split_kernel(const unsigned short* __restrict__ qkv,
                                  const int* __restrict__ pos,
                                  const float* __restrict__ inv_freq,
                                  unsigned short* __restrict__ qh,
                                  unsigned short* __restrict__ kh,
                                  unsigned short* __restrict__ vh) {
  const int s = blockIdx.x, h = blockIdx.y, d2 = threadIdx.x;  // d2 in [0,64)
  float p = (float)pos[s];
  float f = p * inv_freq[d2];
  float sn, c;
  __sincosf(f, &sn, &c);
  const size_t qo = (size_t)s * QKVN + (size_t)h * HDIM;
  float q1 = bf2f(qkv[qo + d2]),            q2 = bf2f(qkv[qo + 64 + d2]);
  float k1 = bf2f(qkv[qo + DM + d2]),       k2 = bf2f(qkv[qo + DM + 64 + d2]);
  const float scale = 0.08838834764831845f;  // 128^-0.5
  const size_t ho = (size_t)h * SEQ * HDIM + (size_t)s * HDIM;
  qh[ho + d2]      = f2bf((q1 * c - q2 * sn) * scale);
  qh[ho + 64 + d2] = f2bf((q2 * c + q1 * sn) * scale);
  kh[ho + d2]      = f2bf(k1 * c - k2 * sn);
  kh[ho + 64 + d2] = f2bf(k2 * c + k1 * sn);
  vh[ho + d2]      = qkv[qo + 2 * DM + d2];
  vh[ho + 64 + d2] = qkv[qo + 2 * DM + 64 + d2];
}

// ---- tiled WMMA GEMM: C[m,n] = A[m,:]·B[n,:]  (both row-major bf16) --------
// Block tile 128x128, 4 waves in 2x2, each wave register-blocks a 64x64 tile:
// 16 WMMAs per 16 ds_load_b128 per k-step (1:1 tensor-to-LDS issue ratio).
// Stores only rows where mask[row]==expert (f32 or bf16 destination).
__global__ __launch_bounds__(128)
void gemm_bf16_sel(const unsigned short* __restrict__ A, int lda,
                   const unsigned short* __restrict__ B, int ldb,
                   float* __restrict__ Cf, unsigned short* __restrict__ Cb,
                   int ldc, int K, const int* __restrict__ mask, int expert) {
  __shared__ unsigned short As[128 * 32];
  __shared__ unsigned short Bs[128 * 32];
  const int t = threadIdx.x, lane = t & 31, w = t >> 5;
  const int bm = blockIdx.y * 128, bn = blockIdx.x * 128;
  const int wm = (w >> 1) * 64, wn = (w & 1) * 64;

  v8f acc[4][4];
#pragma unroll
  for (int mi = 0; mi < 4; ++mi)
#pragma unroll
    for (int ni = 0; ni < 4; ++ni)
#pragma unroll
      for (int r = 0; r < 8; ++r) acc[mi][ni][r] = 0.f;

  const unsigned short* gA = A + (size_t)(bm + t) * lda;  // thread t owns row t
  const unsigned short* gB = B + (size_t)(bn + t) * ldb;

  for (int k0 = 0; k0 < K; k0 += 32) {
    {  // stage 128x32 A and B tiles: one 64B row per thread each
      const uint4* ga = (const uint4*)(gA + k0);
      const uint4* gb = (const uint4*)(gB + k0);
      uint4* la = (uint4*)(As + t * 32);
      uint4* lb = (uint4*)(Bs + t * 32);
#pragma unroll
      for (int q = 0; q < 4; ++q) { la[q] = ga[q]; lb[q] = gb[q]; }
      // hide HBM latency: prefetch two k-steps ahead (speculative, dropped OOB)
      __builtin_prefetch(gA + k0 + 64, 0, 3);
      __builtin_prefetch(gB + k0 + 64, 0, 3);
    }
    __syncthreads();
    v16bf af[4], bf[4];
#pragma unroll
    for (int mi = 0; mi < 4; ++mi) af[mi] = load_frag_a(As, wm + mi * 16, 32, 0, lane);
#pragma unroll
    for (int ni = 0; ni < 4; ++ni) bf[ni] = load_frag_b(Bs, wn + ni * 16, 32, 0, lane);
#pragma unroll
    for (int mi = 0; mi < 4; ++mi)
#pragma unroll
      for (int ni = 0; ni < 4; ++ni)
        acc[mi][ni] = wmma_bf16(af[mi], bf[ni], acc[mi][ni]);
    __syncthreads();
  }

  const int hf = lane >> 4, c0 = lane & 15;
#pragma unroll
  for (int mi = 0; mi < 4; ++mi)
#pragma unroll
    for (int ni = 0; ni < 4; ++ni)
#pragma unroll
      for (int r = 0; r < 8; ++r) {
        int grow = bm + wm + mi * 16 + r + 8 * hf;
        int gcol = bn + wn + ni * 16 + c0;
        if (mask[grow] == expert) {
          if (Cf) Cf[(size_t)grow * ldc + gcol] = acc[mi][ni][r];
          else    Cb[(size_t)grow * ldc + gcol] = f2bf(acc[mi][ni][r]);
        }
      }
}

// ---- flash attention: one (head, 64-row q block) per workgroup -------------
__global__ __launch_bounds__(128)
void flash_attn_kernel(const unsigned short* __restrict__ qh,
                       const unsigned short* __restrict__ kh,
                       const unsigned short* __restrict__ vh,
                       unsigned short* __restrict__ ctx) {
  __shared__ unsigned short Ks[64 * HDIM];   // K tile, row-major [64][128]
  __shared__ unsigned short Vt[HDIM * 64];   // V tile transposed [128][64]
  __shared__ unsigned short Ps[4 * 16 * 64]; // per-wave P staging
  const int h = blockIdx.x, ib = blockIdx.y;
  const int t = threadIdx.x, lane = t & 31, w = t >> 5;
  const int hf = lane >> 4, c0 = lane & 15;
  const size_t hbase = (size_t)h * SEQ * HDIM;

  // Q fragments for this wave's 16 rows, kept in registers for the whole block
  v16bf qf[4];
  {
    const unsigned short* qb = qh + hbase + (size_t)(ib * 64 + w * 16) * HDIM;
#pragma unroll
    for (int kk = 0; kk < 4; ++kk) qf[kk] = load_frag_a(qb, 0, HDIM, kk * 32, lane);
  }

  v8f o[8];
#pragma unroll
  for (int d = 0; d < 8; ++d)
#pragma unroll
    for (int r = 0; r < 8; ++r) o[d][r] = 0.f;
  float m[8], l[8];
#pragma unroll
  for (int r = 0; r < 8; ++r) { m[r] = -__builtin_inff(); l[r] = 0.f; }

  for (int j = 0; j <= ib; ++j) {
    {  // stage K tile (64x128 bf16), 128B per thread
      int row = t >> 1, seg = t & 1;
      const uint4* g = (const uint4*)(kh + hbase + (size_t)(j * 64 + row) * HDIM + seg * 64);
      uint4* s = (uint4*)(Ks + row * HDIM + seg * 64);
#pragma unroll
      for (int q = 0; q < 8; ++q) s[q] = g[q];
    }
    {  // stage V tile transposed: thread t owns dim d=t, gathers 64 rows
      int d = t;
#pragma unroll
      for (int g8 = 0; g8 < 8; ++g8) {
        union { unsigned short s[8]; uint4 u; } tmp;
#pragma unroll
        for (int e = 0; e < 8; ++e)
          tmp.s[e] = vh[hbase + (size_t)(j * 64 + g8 * 8 + e) * HDIM + d];
        *(uint4*)(Vt + d * 64 + g8 * 8) = tmp.u;
      }
    }
    __syncthreads();

    // S = Q·K^T  (16x64 per wave, f32 accum; scale pre-folded into Q)
    v8f sfr[4];
#pragma unroll
    for (int nf = 0; nf < 4; ++nf) {
#pragma unroll
      for (int r = 0; r < 8; ++r) sfr[nf][r] = 0.f;
#pragma unroll
      for (int kk = 0; kk < 4; ++kk) {
        v16bf bf = load_frag_b(Ks, nf * 16, HDIM, kk * 32, lane);
        sfr[nf] = wmma_bf16(qf[kk], bf, sfr[nf]);
      }
    }
    if (j == ib) {  // causal mask on the diagonal block
#pragma unroll
      for (int nf = 0; nf < 4; ++nf)
#pragma unroll
        for (int r = 0; r < 8; ++r) {
          int qr = w * 16 + r + 8 * hf;
          int kc = nf * 16 + c0;
          if (kc > qr) sfr[nf][r] = -__builtin_inff();
        }
    }

    // online softmax update
    float mnew[8], al[8], psum[8];
#pragma unroll
    for (int r = 0; r < 8; ++r) {
      float v = fmaxf(fmaxf(sfr[0][r], sfr[1][r]), fmaxf(sfr[2][r], sfr[3][r]));
      v = red_max16(v);
      mnew[r] = fmaxf(m[r], v);
      al[r] = __expf(m[r] - mnew[r]);
      l[r] *= al[r];
      psum[r] = 0.f;
      m[r] = mnew[r];
    }
    unsigned short* pw = Ps + w * (16 * 64);
#pragma unroll
    for (int nf = 0; nf < 4; ++nf)
#pragma unroll
      for (int r = 0; r < 8; ++r) {
        float p = __expf(sfr[nf][r] - mnew[r]);
        psum[r] += p;
        pw[(r + 8 * hf) * 64 + nf * 16 + c0] = f2bf(p);  // C-layout -> LDS
      }
#pragma unroll
    for (int r = 0; r < 8; ++r) l[r] += red_sum16(psum[r]);
#pragma unroll
    for (int d = 0; d < 8; ++d)
#pragma unroll
      for (int r = 0; r < 8; ++r) o[d][r] *= al[r];

    // O += P·V  (P reloaded from LDS as A-fragments, Vt as B-fragments)
#pragma unroll
    for (int kk2 = 0; kk2 < 2; ++kk2) {
      v16bf pf = load_frag_a(pw, 0, 64, kk2 * 32, lane);
#pragma unroll
      for (int d = 0; d < 8; ++d) {
        v16bf vf = load_frag_b(Vt, d * 16, 64, kk2 * 32, lane);
        o[d] = wmma_bf16(pf, vf, o[d]);
      }
    }
    __syncthreads();
  }

  // finalize: O / l -> ctx (token-major [S][D], head h occupies cols h*128..)
#pragma unroll
  for (int d = 0; d < 8; ++d)
#pragma unroll
    for (int r = 0; r < 8; ++r) {
      int srow = ib * 64 + w * 16 + r + 8 * hf;
      ctx[(size_t)srow * DM + h * HDIM + d * 16 + c0] = f2bf(o[d][r] / l[r]);
    }
}

// ---- host-side orchestration ----------------------------------------------
extern "C" void kernel_launch(void* const* d_in, const int* in_sizes, int n_in,
                              void* d_out, int out_size, void* d_ws, size_t ws_size,
                              hipStream_t stream) {
  (void)in_sizes; (void)n_in; (void)out_size; (void)ws_size;
  const float* hidden    = (const float*)d_in[0];
  const int*   tt        = (const int*)d_in[1];
  const int*   pos       = (const int*)d_in[2];
  const float* wqkv_vis  = (const float*)d_in[3];
  const float* wqkv_lang = (const float*)d_in[4];
  const float* wo_vis    = (const float*)d_in[5];
  const float* wo_lang   = (const float*)d_in[6];
  const float* inv_freq  = (const float*)d_in[7];

  char* ws = (char*)d_ws;
  size_t off = 0;
  auto take = [&](size_t bytes) -> char* {
    char* p = ws + off;
    off = (off + bytes + 255) & ~(size_t)255;
    return p;
  };
  unsigned short* wqv  = (unsigned short*)take((size_t)QKVN * DM * 2);
  unsigned short* wql  = (unsigned short*)take((size_t)QKVN * DM * 2);
  unsigned short* wov  = (unsigned short*)take((size_t)DM * DM * 2);
  unsigned short* wol  = (unsigned short*)take((size_t)DM * DM * 2);
  unsigned short* hb   = (unsigned short*)take((size_t)SEQ * DM * 2);
  unsigned short* qkvb = (unsigned short*)take((size_t)SEQ * QKVN * 2);
  unsigned short* qhb  = (unsigned short*)take((size_t)NH * SEQ * HDIM * 2);
  unsigned short* khb  = (unsigned short*)take((size_t)NH * SEQ * HDIM * 2);
  unsigned short* vhb  = (unsigned short*)take((size_t)NH * SEQ * HDIM * 2);
  unsigned short* ctxb = (unsigned short*)take((size_t)SEQ * DM * 2);
  int* maskp           = (int*)take(SEQ * sizeof(int));

  // 1) precision down-convert (weights once -> bf16; halves HBM traffic)
  f32_to_bf16_kernel<<<4096, 256, 0, stream>>>(wqkv_vis,  wqv, (long long)QKVN * DM);
  f32_to_bf16_kernel<<<4096, 256, 0, stream>>>(wqkv_lang, wql, (long long)QKVN * DM);
  f32_to_bf16_kernel<<<2048, 256, 0, stream>>>(wo_vis,  wov, (long long)DM * DM);
  f32_to_bf16_kernel<<<2048, 256, 0, stream>>>(wo_lang, wol, (long long)DM * DM);
  f32_to_bf16_kernel<<<1024, 256, 0, stream>>>(hidden,  hb,  (long long)SEQ * DM);
  build_mask_kernel<<<SEQ / 256, 256, 0, stream>>>(tt, maskp);

  // 2) QKV projection: both experts, disjoint conditional row stores
  dim3 gq(QKVN / 128, SEQ / 128);
  gemm_bf16_sel<<<gq, 128, 0, stream>>>(hb, DM, wql, DM, nullptr, qkvb, QKVN, DM, maskp, 0);
  gemm_bf16_sel<<<gq, 128, 0, stream>>>(hb, DM, wqv, DM, nullptr, qkvb, QKVN, DM, maskp, 1);

  // 3) RoPE + head-major split (softmax scale folded into q)
  rope_split_kernel<<<dim3(SEQ, NH), 64, 0, stream>>>(qkvb, pos, inv_freq, qhb, khb, vhb);

  // 4) causal flash attention
  flash_attn_kernel<<<dim3(NH, SEQ / 64), 128, 0, stream>>>(qhb, khb, vhb, ctxb);

  // 5) output projection: both experts, conditional row stores into d_out (f32)
  dim3 go(DM / 128, SEQ / 128);
  gemm_bf16_sel<<<go, 128, 0, stream>>>(ctxb, DM, wol, DM, (float*)d_out, nullptr, DM, DM, maskp, 0);
  gemm_bf16_sel<<<go, 128, 0, stream>>>(ctxb, DM, wov, DM, (float*)d_out, nullptr, DM, DM, maskp, 1);
}
